// MoeLayer_18683107737665
// MI455X (gfx1250) — compile-verified, matrix operand order checked by reading, not run on hardware
//
#include <hip/hip_runtime.h>

#define T_TOK 8192   // B*S
#define DD    1024
#define EE    8
#define KSEL  2
#define HH    4096

typedef __bf16 v16bf __attribute__((ext_vector_type(16)));
typedef float  v8f   __attribute__((ext_vector_type(8)));

__device__ inline unsigned short f2bf(float f) {
  unsigned int u = __float_as_uint(f);
  u += 0x7FFFu + ((u >> 16) & 1u);   // round-to-nearest-even
  return (unsigned short)(u >> 16);
}
// gelu(v) = v * sigmoid(2 * 0.79788456*(v + 0.044715 v^3)); one v_exp + one v_rcp
__device__ inline float gelu_fast(float v) {
  float u = v + 0.044715f * v * v * v;
  float s = __builtin_amdgcn_rcpf(1.f + __expf(-1.5957691216057308f * u));
  return v * s;
}

// CDNA5 async memory->LDS copy (no VGPR round trip, tracked by ASYNCcnt)
__device__ inline void async_copy_b128(unsigned lds_off, const void* gaddr) {
  asm volatile("global_load_async_to_lds_b128 %0, %1, off"
               :: "v"(lds_off), "v"(gaddr) : "memory");
}
__device__ inline void wait_async0() {
#if __has_builtin(__builtin_amdgcn_s_wait_asynccnt)
  __builtin_amdgcn_s_wait_asynccnt(0);
#else
  asm volatile("s_wait_asynccnt 0x0" ::: "memory");
#endif
}
__device__ inline unsigned lds_off_of(const void* p) {
  return (unsigned)(uintptr_t)p;     // LDS aperture is 4GB-aligned: low 32 bits = LDS offset
}

// ---------------- small utility kernels ----------------
__global__ void init_counts_kernel(int* counts) {
  if (threadIdx.x < EE) counts[threadIdx.x] = 0;
}

__global__ void zero_f4_kernel(float4* p, int n4) {
  int i = blockIdx.x * blockDim.x + threadIdx.x;
  if (i < n4) p[i] = make_float4(0.f, 0.f, 0.f, 0.f);
}

__global__ void cvt_bf16_kernel(const float* __restrict__ src,
                                unsigned short* __restrict__ dst, int n4) {
  int i = blockIdx.x * blockDim.x + threadIdx.x;
  if (i >= n4) return;
  float4 v = ((const float4*)src)[i];
  ushort4 o;
  o.x = f2bf(v.x); o.y = f2bf(v.y); o.z = f2bf(v.z); o.w = f2bf(v.w);
  ((ushort4*)dst)[i] = o;
}

// transpose + convert: src [Z][R][C] f32 -> dst [Z][C][R] bf16  (32x32 tiles)
__global__ __launch_bounds__(256) void transpose_cvt_kernel(
    const float* __restrict__ src, unsigned short* __restrict__ dst, int R, int C) {
  __shared__ unsigned short tile[32][33];
  const float* s = src + (size_t)blockIdx.z * R * C;
  unsigned short* d = dst + (size_t)blockIdx.z * R * C;
  const int c0 = blockIdx.x * 32, r0 = blockIdx.y * 32;
  const int tx = threadIdx.x & 31, ty = threadIdx.x >> 5;   // 32 x 8
#pragma unroll
  for (int j = 0; j < 32; j += 8)
    tile[ty + j][tx] = f2bf(s[(size_t)(r0 + ty + j) * C + c0 + tx]);
  __syncthreads();
#pragma unroll
  for (int j = 0; j < 32; j += 8)
    d[(size_t)(c0 + ty + j) * R + r0 + tx] = tile[tx][ty + j];
}

// ---------------- gating: one wave32 per token ----------------
__global__ __launch_bounds__(256) void gate_kernel(
    const float* __restrict__ x, const float* __restrict__ gw,
    const float* __restrict__ gb, int* __restrict__ counts,
    int* __restrict__ sel, float* __restrict__ selwt) {
  const int lane  = threadIdx.x & 31;
  const int token = blockIdx.x * 8 + (threadIdx.x >> 5);
  float acc[EE];
#pragma unroll
  for (int e = 0; e < EE; ++e) acc[e] = 0.f;
  const float* xr = x + (size_t)token * DD;
  for (int d = lane; d < DD; d += 32) {
    float xv = xr[d];
    const float4* g = (const float4*)(gw + (size_t)d * EE);
    float4 g0 = g[0], g1 = g[1];
    acc[0] += xv * g0.x; acc[1] += xv * g0.y; acc[2] += xv * g0.z; acc[3] += xv * g0.w;
    acc[4] += xv * g1.x; acc[5] += xv * g1.y; acc[6] += xv * g1.z; acc[7] += xv * g1.w;
  }
#pragma unroll
  for (int e = 0; e < EE; ++e)
    for (int off = 16; off > 0; off >>= 1) acc[e] += __shfl_xor(acc[e], off, 32);
  if (lane == 0) {
    float l[EE];
#pragma unroll
    for (int e = 0; e < EE; ++e) l[e] = acc[e] + gb[e];
    int i0 = 0;
    for (int e = 1; e < EE; ++e) if (l[e] > l[i0]) i0 = e;       // ties -> lower idx
    int i1 = (i0 == 0) ? 1 : 0;
    for (int e = 0; e < EE; ++e) if (e != i0 && l[e] > l[i1]) i1 = e;
    float e1 = __expf(l[i1] - l[i0]);     // l[i1] <= l[i0], safe
    float inv = 1.f / (1.f + e1);
    sel[2 * token]     = i0;  selwt[2 * token]     = inv;
    sel[2 * token + 1] = i1;  selwt[2 * token + 1] = e1 * inv;
    atomicAdd(&counts[i0], 1);
    atomicAdd(&counts[i1], 1);
  }
}

__global__ void scan_kernel(const int* counts, int* offsets, int* cursors) {
  if (threadIdx.x == 0 && blockIdx.x == 0) {
    int o = 0;
    for (int e = 0; e < EE; ++e) { offsets[e] = o; cursors[e] = o; o += counts[e]; }
    offsets[EE] = o;
  }
}

__global__ void scatter_kernel(const int* __restrict__ sel, const float* __restrict__ selwt,
                               int* __restrict__ cursors, int* __restrict__ rows,
                               float* __restrict__ rowwt) {
  int t = blockIdx.x * blockDim.x + threadIdx.x;
  if (t >= T_TOK) return;
#pragma unroll
  for (int k = 0; k < KSEL; ++k) {
    int e = sel[2 * t + k];
    int slot = atomicAdd(&cursors[e], 1);
    rows[slot]  = t;
    rowwt[slot] = selwt[2 * t + k];
  }
}

// ---------------- FFN1: h = gelu(x_gathered @ w1[e] + b1[e]) -> bf16 ----------------
// block tile 128(M) x 64(N), K step 32, 8 waves each 32x32 (2x2 WMMA tiles).
// Double-buffered async global->LDS staging overlaps HBM copy with WMMA compute.
__global__ __launch_bounds__(256) void ffn1_kernel(
    const unsigned short* __restrict__ xb, const unsigned short* __restrict__ w1t,
    const float* __restrict__ b1, const int* __restrict__ rows,
    const int* __restrict__ offsets, unsigned short* __restrict__ hbuf) {
  const int e   = blockIdx.z;
  const int off = offsets[e];
  const int cnt = offsets[e + 1] - off;
  const int m0  = blockIdx.x * 128;
  if (m0 >= cnt) return;
  const int n0  = blockIdx.y * 64;

  __shared__ alignas(16) unsigned short As[2][128][32];  // [buf][m][k]
  __shared__ alignas(16) unsigned short Bs[2][64][32];   // [buf][n][k]
  __shared__ int tok[128];

  const int tid = threadIdx.x;
  if (tid < 128) {
    int r = m0 + tid;
    tok[tid] = rows[off + (r < cnt ? r : 0)];
  }
  __syncthreads();

  const unsigned short* wB = w1t + (size_t)e * DD * HH;

  const int wave  = tid >> 5, lane = tid & 31;
  const int wm    = (wave & 3) * 32;
  const int wn    = (wave >> 2) * 32;
  const int lhalf = lane >> 4, lidx = lane & 15;
  const int rowc  = tid >> 2, kc = (tid & 3) * 8;     // per-thread 16B chunk coords

  auto stage = [&](int buf, int k0) {
    async_copy_b128(lds_off_of(&As[buf][rowc][kc]),
                    xb + (size_t)tok[rowc] * DD + k0 + kc);
    async_copy_b128(lds_off_of(&As[buf][64 + rowc][kc]),
                    xb + (size_t)tok[64 + rowc] * DD + k0 + kc);
    async_copy_b128(lds_off_of(&Bs[buf][rowc][kc]),
                    wB + (size_t)(n0 + rowc) * DD + k0 + kc);
  };

  const v8f vz = {0.f, 0.f, 0.f, 0.f, 0.f, 0.f, 0.f, 0.f};
  v8f acc[2][2] = {{vz, vz}, {vz, vz}};

  const int nk = DD / 32;
  stage(0, 0);
  for (int it = 0; it < nk; ++it) {
    const int cur = it & 1;
    wait_async0();
    __syncthreads();
    if (it + 1 < nk) stage(1 - cur, (it + 1) * 32);   // prefetch next tile during compute

    v16bf af[2], bfg[2];
#pragma unroll
    for (int t = 0; t < 2; ++t) {
      union { v16bf v; uint4 q[2]; } ua;
      ua.q[0] = *(const uint4*)&As[cur][wm + t * 16 + lidx][lhalf * 8];
      ua.q[1] = *(const uint4*)&As[cur][wm + t * 16 + lidx][16 + lhalf * 8];
      af[t] = ua.v;
      union { v16bf v; uint4 q[2]; } ub2;
      ub2.q[0] = *(const uint4*)&Bs[cur][wn + t * 16 + lidx][lhalf * 8];
      ub2.q[1] = *(const uint4*)&Bs[cur][wn + t * 16 + lidx][16 + lhalf * 8];
      bfg[t] = ub2.v;
    }
#pragma unroll
    for (int tm = 0; tm < 2; ++tm)
#pragma unroll
      for (int tn = 0; tn < 2; ++tn)
        acc[tm][tn] = __builtin_amdgcn_wmma_f32_16x16x32_bf16(
            false, af[tm], false, bfg[tn], (short)0, acc[tm][tn], false, false);
  }

#pragma unroll
  for (int tm = 0; tm < 2; ++tm)
#pragma unroll
    for (int tn = 0; tn < 2; ++tn) {
      int gn = n0 + wn + tn * 16 + lidx;
      float bias = b1[(size_t)e * HH + gn];
#pragma unroll
      for (int i = 0; i < 8; ++i) {
        int gm = m0 + wm + tm * 16 + lhalf * 8 + i;
        if (gm < cnt) {
          float v = acc[tm][tn][i] + bias;
          hbuf[(size_t)(off + gm) * HH + gn] = f2bf(gelu_fast(v));
        }
      }
    }
}

// ---------------- FFN2: out[token] += wt * (h @ w2[e] + b2[e]) ----------------
__global__ __launch_bounds__(256) void ffn2_kernel(
    const unsigned short* __restrict__ hbuf, const unsigned short* __restrict__ w2t,
    const float* __restrict__ b2, const int* __restrict__ rows,
    const float* __restrict__ rowwt, const int* __restrict__ offsets,
    float* __restrict__ out) {
  const int e   = blockIdx.z;
  const int off = offsets[e];
  const int cnt = offsets[e + 1] - off;
  const int m0  = blockIdx.x * 128;
  if (m0 >= cnt) return;
  const int n0  = blockIdx.y * 64;

  __shared__ alignas(16) unsigned short As[2][128][32];
  __shared__ alignas(16) unsigned short Bs[2][64][32];

  const int tid = threadIdx.x;
  const unsigned short* wB = w2t + (size_t)e * HH * DD;

  const int wave  = tid >> 5, lane = tid & 31;
  const int wm    = (wave & 3) * 32;
  const int wn    = (wave >> 2) * 32;
  const int lhalf = lane >> 4, lidx = lane & 15;
  const int rowc  = tid >> 2, kc = (tid & 3) * 8;

  // clamped hbuf row indices (stay inside valid region; extra rows never stored)
  const int ra = (m0 + rowc      < cnt) ? m0 + rowc      : cnt - 1;
  const int rb = (m0 + 64 + rowc < cnt) ? m0 + 64 + rowc : cnt - 1;

  auto stage = [&](int buf, int k0) {
    async_copy_b128(lds_off_of(&As[buf][rowc][kc]),
                    hbuf + (size_t)(off + ra) * HH + k0 + kc);
    async_copy_b128(lds_off_of(&As[buf][64 + rowc][kc]),
                    hbuf + (size_t)(off + rb) * HH + k0 + kc);
    async_copy_b128(lds_off_of(&Bs[buf][rowc][kc]),
                    wB + (size_t)(n0 + rowc) * HH + k0 + kc);
  };

  const v8f vz = {0.f, 0.f, 0.f, 0.f, 0.f, 0.f, 0.f, 0.f};
  v8f acc[2][2] = {{vz, vz}, {vz, vz}};

  const int nk = HH / 32;
  stage(0, 0);
  for (int it = 0; it < nk; ++it) {
    const int cur = it & 1;
    wait_async0();
    __syncthreads();
    if (it + 1 < nk) stage(1 - cur, (it + 1) * 32);

    v16bf af[2], bfg[2];
#pragma unroll
    for (int t = 0; t < 2; ++t) {
      union { v16bf v; uint4 q[2]; } ua;
      ua.q[0] = *(const uint4*)&As[cur][wm + t * 16 + lidx][lhalf * 8];
      ua.q[1] = *(const uint4*)&As[cur][wm + t * 16 + lidx][16 + lhalf * 8];
      af[t] = ua.v;
      union { v16bf v; uint4 q[2]; } ub2;
      ub2.q[0] = *(const uint4*)&Bs[cur][wn + t * 16 + lidx][lhalf * 8];
      ub2.q[1] = *(const uint4*)&Bs[cur][wn + t * 16 + lidx][16 + lhalf * 8];
      bfg[t] = ub2.v;
    }
#pragma unroll
    for (int tm = 0; tm < 2; ++tm)
#pragma unroll
      for (int tn = 0; tn < 2; ++tn)
        acc[tm][tn] = __builtin_amdgcn_wmma_f32_16x16x32_bf16(
            false, af[tm], false, bfg[tn], (short)0, acc[tm][tn], false, false);
  }

#pragma unroll
  for (int tm = 0; tm < 2; ++tm)
#pragma unroll
    for (int tn = 0; tn < 2; ++tn) {
      int gn = n0 + wn + tn * 16 + lidx;
      float bias = b2[(size_t)e * DD + gn];
#pragma unroll
      for (int i = 0; i < 8; ++i) {
        int gm = m0 + wm + tm * 16 + lhalf * 8 + i;
        if (gm < cnt) {
          int grow = off + gm;
          int token = rows[grow];
          float wt  = rowwt[grow];
          float v = acc[tm][tn][i] + bias;
          atomicAdd(out + (size_t)token * DD + gn, wt * v);
        }
      }
    }
}

// ---------------- launch ----------------
extern "C" void kernel_launch(void* const* d_in, const int* in_sizes, int n_in,
                              void* d_out, int out_size, void* d_ws, size_t ws_size,
                              hipStream_t stream) {
  (void)in_sizes; (void)n_in; (void)out_size; (void)ws_size;
  const float* x      = (const float*)d_in[0];
  const float* gate_w = (const float*)d_in[1];
  const float* gate_b = (const float*)d_in[2];
  const float* w1     = (const float*)d_in[3];
  const float* b1     = (const float*)d_in[4];
  const float* w2     = (const float*)d_in[5];
  const float* b2     = (const float*)d_in[6];
  float* out = (float*)d_out;

  // workspace layout (~286 MB total)
  char* ws = (char*)d_ws;
  size_t o = 0;
  auto take = [&](size_t bytes) { char* p = ws + o; o = (o + bytes + 255) & ~(size_t)255; return p; };
  unsigned short* xb   = (unsigned short*)take((size_t)T_TOK * DD * 2);
  unsigned short* w1t  = (unsigned short*)take((size_t)EE * DD * HH * 2);  // [E][H][D] bf16
  unsigned short* w2t  = (unsigned short*)take((size_t)EE * HH * DD * 2);  // [E][D][H] bf16
  unsigned short* hbuf = (unsigned short*)take((size_t)2 * T_TOK * HH * 2);
  int*   counts  = (int*)take(EE * 4);
  int*   offsets = (int*)take((EE + 1) * 4);
  int*   cursors = (int*)take(EE * 4);
  int*   sel     = (int*)take((size_t)T_TOK * 2 * 4);
  float* selwt   = (float*)take((size_t)T_TOK * 2 * 4);
  int*   rowsA   = (int*)take((size_t)2 * T_TOK * 4);
  float* rowwt   = (float*)take((size_t)2 * T_TOK * 4);

  init_counts_kernel<<<1, 32, 0, stream>>>(counts);
  {
    int n4 = T_TOK * DD / 4;
    zero_f4_kernel<<<(n4 + 255) / 256, 256, 0, stream>>>((float4*)out, n4);
    cvt_bf16_kernel<<<(n4 + 255) / 256, 256, 0, stream>>>(x, xb, n4);
  }
  // w1 [E][D][H] -> w1t [E][H][D]; w2 [E][H][D] -> w2t [E][D][H]
  transpose_cvt_kernel<<<dim3(HH / 32, DD / 32, EE), 256, 0, stream>>>(w1, w1t, DD, HH);
  transpose_cvt_kernel<<<dim3(DD / 32, HH / 32, EE), 256, 0, stream>>>(w2, w2t, HH, DD);

  gate_kernel<<<T_TOK / 8, 256, 0, stream>>>(x, gate_w, gate_b, counts, sel, selwt);
  scan_kernel<<<1, 1, 0, stream>>>(counts, offsets, cursors);
  scatter_kernel<<<(T_TOK + 255) / 256, 256, 0, stream>>>(sel, selwt, cursors, rowsA, rowwt);
  ffn1_kernel<<<dim3(T_TOK / 128, HH / 64, EE), 256, 0, stream>>>(xb, w1t, b1, rowsA, offsets, hbuf);
  ffn2_kernel<<<dim3(T_TOK / 128, DD / 64, EE), 256, 0, stream>>>(hbuf, w2t, b2, rowsA, rowwt, offsets, out);
}
